// SequenceContextEncoder_57019985822039
// MI455X (gfx1250) — matrix operand order repaired and checked by reading, not compiled, not run on hardware
//
#include <hip/hip_runtime.h>

// ---------------------------------------------------------------------------
// SequenceContextEncoder for MI455X (gfx1250, wave32, WMMA).
// B=2, T=4096, D=256, H=8, HD=32, L=4, FF=1024, SPLIT read on device.
// All matmuls on v_wmma_f32_16x16x32_f16; every fragment fed by b128 loads:
//   - weights pre-packed into per-lane-contiguous WMMA fragment order
//   - V stored transposed [B,H,HD,T] so P@V B-fragments are contiguous
//   - A / K^T / LDS-P fragments loaded as 2 x 16-byte vectors per lane
// ---------------------------------------------------------------------------

#define BB   2
#define TT   4096
#define DD   256
#define HH   8
#define HDD  32
#define LL   4
#define FFD  1024
#define MM   (BB * TT)     // 8192 rows

typedef __attribute__((ext_vector_type(16))) _Float16 v16h;
typedef __attribute__((ext_vector_type(8)))  _Float16 v8h;
typedef __attribute__((ext_vector_type(8)))  float    v8f;

// ---------------------------------------------------------------------------
// WMMA helpers (CDNA5 VGPR layouts, ISA 7.12.2)
// ---------------------------------------------------------------------------

__device__ __forceinline__ v8f wmma_f16(v16h a, v16h b, v8f c) {
  return __builtin_amdgcn_wmma_f32_16x16x32_f16(
      false, a, false, b, (short)0, c, false, false);
}

__device__ __forceinline__ v16h cat8(v8h lo, v8h hi) {
  return __builtin_shufflevector(lo, hi, 0, 1, 2, 3, 4, 5, 6, 7,
                                 8, 9, 10, 11, 12, 13, 14, 15);
}

// A fragment: 16x32 f16 tile, row-major, lda elements (lda % 8 == 0, 16B rows).
// Lane layout: element e -> K = hf*16+e.
// => two contiguous 8-half runs per lane: 2 x b128 loads.
__device__ __forceinline__ v16h load_a16x32(const _Float16* __restrict__ A, int lda) {
  const int lane = threadIdx.x & 31;
  const _Float16* p = A + (size_t)(lane & 15) * lda + ((lane >> 4) << 4);
  return cat8(*(const v8h*)p, *(const v8h*)(p + 8));
}

// Packed B fragment: one 32x16 tile = 512 halves, lane's 16 values contiguous.
__device__ __forceinline__ v16h load_b_packed(const _Float16* __restrict__ P) {
  const int lane = threadIdx.x & 31;
  const _Float16* p = P + lane * 16;
  return cat8(*(const v8h*)p, *(const v8h*)(p + 8));
}

// B fragment from transposed source: logical B[k][n] = S[n*ldn + k].
// Lane: n = lane&15, K = hf*16 + e -> 16 contiguous halves: 2 x b128 loads.
__device__ __forceinline__ v16h load_bT32x16(const _Float16* __restrict__ S, int ldn) {
  const int lane = threadIdx.x & 31;
  const _Float16* p = S + (size_t)(lane & 15) * ldn + ((lane >> 4) << 4);
  return cat8(*(const v8h*)p, *(const v8h*)(p + 8));
}

// ---------------------------------------------------------------------------
// Small helpers
// ---------------------------------------------------------------------------

__device__ __forceinline__ float gelu_tanh(float x) {
  const float c0 = 0.7978845608028654f;  // sqrt(2/pi)
  float t = tanhf(c0 * (x + 0.044715f * x * x * x));
  return 0.5f * x * (1.0f + t);
}

__device__ __forceinline__ float hgrp_max(float v) {
#pragma unroll
  for (int m = 1; m <= 8; m <<= 1) v = fmaxf(v, __shfl_xor(v, m, 32));
  return v;
}
__device__ __forceinline__ float hgrp_sum(float v) {
#pragma unroll
  for (int m = 1; m <= 8; m <<= 1) v += __shfl_xor(v, m, 32);
  return v;
}
__device__ __forceinline__ float wave_sum(float v) {
#pragma unroll
  for (int m = 1; m <= 16; m <<= 1) v += __shfl_xor(v, m, 32);
  return v;
}

// ---------------------------------------------------------------------------
// Kernels
// ---------------------------------------------------------------------------

// Pack W[L][K][N] f32 into WMMA-fragment-ordered f16 tiles:
// dst[(((l*KT + kt)*NT + nt)*32 + lane)*16 + e] =
//     W[l][kt*32 + (lane>>4)*16 + e][nt*16 + (lane&15)]
__global__ void pack_w_kernel(const float* __restrict__ W,
                              _Float16* __restrict__ dst,
                              int K, int N, int total /* L*K*N */) {
  int o = blockIdx.x * blockDim.x + threadIdx.x;
  if (o >= total) return;
  const int e    = o & 15;
  const int lane = (o >> 4) & 31;
  const int tile = o >> 9;
  const int NT = N >> 4;
  const int nt  = tile % NT;
  const int ktl = tile / NT;          // l*KT + kt
  const int KT = K >> 5;
  const int kt = ktl % KT;
  const int l  = ktl / KT;
  const int k = kt * 32 + ((lane >> 4) << 4) + e;
  const int n = nt * 16 + (lane & 15);
  dst[o] = (_Float16)W[((size_t)l * K + k) * N + n];
}

// x = rows; x[:, :split, :] += tte ; write f32 master + f16 copy
__global__ void add_embed_kernel(const float* __restrict__ rows,
                                 const float* __restrict__ tte,
                                 const int* __restrict__ split_p,
                                 float* __restrict__ x32,
                                 _Float16* __restrict__ x16) {
  int i = blockIdx.x * blockDim.x + threadIdx.x;   // over B*T*D
  int d = i % DD;
  int t = (i / DD) % TT;
  int b = i / (DD * TT);
  int split = *split_p;
  float v = rows[i];
  if (t < split) v += tte[((size_t)b * split + t) * DD + d];
  x32[i] = v;
  x16[i] = (_Float16)v;
}

// C[M x N] = A[M x K] @ W[K x N] + bias; W pre-packed (layer base pointer).
// One wave owns a 16x64 strip (A fragment reused across 4 N-subtiles).
// Epilogue: optional GELU, optional f32 residual; outputs: f16 row-major,
// f32 row-major, and/or f16 transposed-to-[B,H,HD,T] (for V).
__global__ void gemm_wmma_kernel(const _Float16* __restrict__ A,
                                 const _Float16* __restrict__ Wp,
                                 const float* __restrict__ bias,
                                 const float* __restrict__ resid,   // nullable
                                 _Float16* __restrict__ outh,       // nullable
                                 float* __restrict__ outf,          // nullable
                                 _Float16* __restrict__ out_vt,     // nullable
                                 int M, int N, int K, int do_gelu) {
  const int wave   = (blockIdx.x * blockDim.x + threadIdx.x) >> 5;
  const int lane   = threadIdx.x & 31;
  const int tilesN = N >> 6;                 // N / 64
  const int m0 = (wave / tilesN) << 4;
  const int n0 = (wave % tilesN) << 6;
  if (m0 >= M) return;
  const int NT = N >> 4;
  const int nt0 = n0 >> 4;

  v8f acc[4] = {};
  for (int k0 = 0; k0 < K; k0 += 32) {
    v16h a = load_a16x32(A + (size_t)m0 * K + k0, K);
    const _Float16* tb = Wp + (((size_t)(k0 >> 5) * NT + nt0) << 9);
#pragma unroll
    for (int t = 0; t < 4; ++t) {
      v16h b = load_b_packed(tb + ((size_t)t << 9));
      acc[t] = wmma_f16(a, b, acc[t]);
    }
  }

  const int hf = lane >> 4, n = lane & 15;
#pragma unroll
  for (int t = 0; t < 4; ++t) {
    const int col = n0 + t * 16 + n;
    const float bs = bias[col];
#pragma unroll
    for (int r = 0; r < 8; ++r) {
      const int row = m0 + hf * 8 + r;
      float v = acc[t][r] + bs;
      if (do_gelu) v = gelu_tanh(v);
      if (resid) v += resid[(size_t)row * N + col];
      if (outf) outf[(size_t)row * N + col] = v;
      if (outh) outh[(size_t)row * N + col] = (_Float16)v;
      if (out_vt) {
        const int bb = row / TT, tok = row % TT;
        const int hh = col >> 5, hd = col & 31;   // HD == 32
        out_vt[(((size_t)bb * HH + hh) * HDD + hd) * TT + tok] = (_Float16)v;
      }
    }
  }
}

// Flash attention with "train keys OR diagonal" mask.
// One wave per (b, h, 16-query tile); Q/K layout [B,T,H,HD] f16, V transposed
// [B,H,HD,T] f16 so all B-fragments are per-lane contiguous (2 x b128).
__global__ void attn_kernel(const _Float16* __restrict__ Q,
                            const _Float16* __restrict__ K,
                            const _Float16* __restrict__ Vt,
                            const int* __restrict__ split_p,
                            _Float16* __restrict__ O) {
  __shared__ __attribute__((aligned(64))) _Float16 plds[8][16 * 32];

  const int wib  = threadIdx.x >> 5;
  const int wave = blockIdx.x * 8 + wib;
  const int lane = threadIdx.x & 31;
  const int qtiles = TT / 16;
  const int b  = wave / (HH * qtiles);
  const int rm = wave % (HH * qtiles);
  const int h  = rm / qtiles;
  const int q0 = (rm % qtiles) * 16;
  const int split = *split_p;
  const float scale = 0.17677669529663687f;   // 1/sqrt(32)

  const int hf = lane >> 4, n = lane & 15;

  // Q fragment: 16 queries x HD(=32), row stride = H*HD = D
  const _Float16* Qb = Q + ((size_t)(b * TT + q0) * HH + h) * HDD;
  const v16h qa = load_a16x32(Qb, DD);

  // per-head V^T plane: [HD][T]
  const _Float16* Vh = Vt + ((size_t)b * HH + h) * HDD * TT;

  float mrow[8], lrow[8];
#pragma unroll
  for (int r = 0; r < 8; ++r) { mrow[r] = -1e30f; lrow[r] = 0.0f; }
  v8f acc0 = {}, acc1 = {};

  for (int kb = 0; kb < TT; kb += 32) {
    // ---- S = (Q @ K^T) for 32 keys: two 16x16 logits tiles ----
    const _Float16* Kb = K + ((size_t)(b * TT + kb) * HH + h) * HDD;
    v16h bk0 = load_bT32x16(Kb, DD);
    v16h bk1 = load_bT32x16(Kb + (size_t)16 * DD, DD);
    v8f s0 = {}, s1 = {};
    s0 = wmma_f16(qa, bk0, s0);
    s1 = wmma_f16(qa, bk1, s1);

    const int key0 = kb + n, key1 = kb + 16 + n;
#pragma unroll
    for (int r = 0; r < 8; ++r) {
      const int qrow = q0 + hf * 8 + r;
      float v0 = (key0 < split || key0 == qrow) ? s0[r] * scale : -1e9f;
      float v1 = (key1 < split || key1 == qrow) ? s1[r] * scale : -1e9f;
      float mx   = hgrp_max(fmaxf(v0, v1));
      float newm = fmaxf(mrow[r], mx);
      float alpha = __expf(mrow[r] - newm);
      float p0 = __expf(v0 - newm);
      float p1 = __expf(v1 - newm);
      float rs = hgrp_sum(p0 + p1);
      lrow[r] = lrow[r] * alpha + rs;
      mrow[r] = newm;
      acc0[r] *= alpha;
      acc1[r] *= alpha;
      const int row = hf * 8 + r;
      plds[wib][row * 32 + n]      = (_Float16)p0;
      plds[wib][row * 32 + 16 + n] = (_Float16)p1;
    }

    // wave-local LDS RAW fence (CDNA5 split counters)
    asm volatile("s_wait_dscnt 0x0" ::: "memory");

    // reload P as A-fragment: two contiguous 8-half runs -> 2 x ds b128
    const _Float16* pr = &plds[wib][(lane & 15) * 32 + ((lane >> 4) << 4)];
    v16h pa = cat8(*(const v8h*)pr, *(const v8h*)(pr + 8));

    // ---- acc += P @ V : V^T plane gives contiguous per-lane loads ----
    const _Float16* Vb = Vh + kb;
    v16h bv0 = load_bT32x16(Vb, TT);                       // hd 0..15
    v16h bv1 = load_bT32x16(Vb + (size_t)16 * TT, TT);     // hd 16..31
    acc0 = wmma_f16(pa, bv0, acc0);
    acc1 = wmma_f16(pa, bv1, acc1);
  }

  // ---- finalize: out = acc / rowsum, store f16 [B,T,H,HD] ----
#pragma unroll
  for (int r = 0; r < 8; ++r) {
    const int qrow = q0 + hf * 8 + r;
    const float inv = 1.0f / lrow[r];
    _Float16* Ob = O + ((size_t)(b * TT + qrow) * HH + h) * HDD;
    Ob[n]      = (_Float16)(acc0[r] * inv);
    Ob[16 + n] = (_Float16)(acc1[r] * inv);
  }
}

// LayerNorm over D=256: one wave per row, 8 values per lane.
__global__ void ln_kernel(const float* __restrict__ in,
                          const float* __restrict__ sc,
                          const float* __restrict__ bi,
                          float* __restrict__ xo32,
                          _Float16* __restrict__ xo16) {
  const int row  = (blockIdx.x * blockDim.x + threadIdx.x) >> 5;
  const int lane = threadIdx.x & 31;
  const float* p = in + (size_t)row * DD;

  float vals[8];
  float s = 0.0f;
#pragma unroll
  for (int i = 0; i < 8; ++i) { vals[i] = p[lane + 32 * i]; s += vals[i]; }
  s = wave_sum(s);
  const float mu = s * (1.0f / DD);

  float vs = 0.0f;
#pragma unroll
  for (int i = 0; i < 8; ++i) { float d = vals[i] - mu; vs += d * d; }
  vs = wave_sum(vs);
  const float rstd = rsqrtf(vs * (1.0f / DD) + 1e-5f);

#pragma unroll
  for (int i = 0; i < 8; ++i) {
    const int col = lane + 32 * i;
    float y = (vals[i] - mu) * rstd * sc[col] + bi[col];
    xo32[(size_t)row * DD + col] = y;
    xo16[(size_t)row * DD + col] = (_Float16)y;
  }
}

__global__ void copy_out_kernel(const float* __restrict__ src,
                                float* __restrict__ dst, int n) {
  int i = blockIdx.x * blockDim.x + threadIdx.x;
  if (i < n) dst[i] = src[i];
}

// ---------------------------------------------------------------------------
// Host orchestration (graph-capture safe: only async launches on `stream`)
// ---------------------------------------------------------------------------

extern "C" void kernel_launch(void* const* d_in, const int* in_sizes, int n_in,
                              void* d_out, int out_size, void* d_ws, size_t ws_size,
                              hipStream_t stream) {
  (void)in_sizes; (void)n_in; (void)out_size; (void)ws_size;

  const float* rows = (const float*)d_in[0];
  const float* tte  = (const float*)d_in[1];
  const float* Wq   = (const float*)d_in[2];
  const float* bq   = (const float*)d_in[3];
  const float* Wk   = (const float*)d_in[4];
  const float* bk   = (const float*)d_in[5];
  const float* Wv   = (const float*)d_in[6];
  const float* bv   = (const float*)d_in[7];
  const float* Wo   = (const float*)d_in[8];
  const float* bo   = (const float*)d_in[9];
  const float* W1   = (const float*)d_in[10];
  const float* b1   = (const float*)d_in[11];
  const float* W2   = (const float*)d_in[12];
  const float* b2   = (const float*)d_in[13];
  const float* ln1s = (const float*)d_in[14];
  const float* ln1b = (const float*)d_in[15];
  const float* ln2s = (const float*)d_in[16];
  const float* ln2b = (const float*)d_in[17];
  const int* split_p = (const int*)d_in[18];

  // ---- workspace carve (256B aligned) ----
  char* w = (char*)d_ws;
  size_t off = 0;
  auto carve = [&](size_t bytes) -> void* {
    void* p = w + off;
    off += (bytes + 255) & ~(size_t)255;
    return p;
  };
  float*    x32   = (float*)   carve((size_t)MM * DD * 4);
  _Float16* x16   = (_Float16*)carve((size_t)MM * DD * 2);
  _Float16* q16   = (_Float16*)carve((size_t)MM * DD * 2);
  _Float16* k16   = (_Float16*)carve((size_t)MM * DD * 2);
  _Float16* vt16  = (_Float16*)carve((size_t)MM * DD * 2);   // [B,H,HD,T]
  _Float16* o16   = (_Float16*)carve((size_t)MM * DD * 2);
  float*    res32 = (float*)   carve((size_t)MM * DD * 4);
  _Float16* h16   = (_Float16*)carve((size_t)MM * FFD * 2);
  _Float16* wq16  = (_Float16*)carve((size_t)LL * DD * DD * 2);
  _Float16* wk16  = (_Float16*)carve((size_t)LL * DD * DD * 2);
  _Float16* wv16  = (_Float16*)carve((size_t)LL * DD * DD * 2);
  _Float16* wo16  = (_Float16*)carve((size_t)LL * DD * DD * 2);
  _Float16* w116  = (_Float16*)carve((size_t)LL * DD * FFD * 2);
  _Float16* w216  = (_Float16*)carve((size_t)LL * FFD * DD * 2);

  const int nWdd = LL * DD * DD;    // 262144
  const int nWff = LL * DD * FFD;   // 1048576

  // ---- weight packing: f32 -> fragment-ordered f16 tiles ----
  pack_w_kernel<<<nWdd / 256, 256, 0, stream>>>(Wq, wq16, DD, DD, nWdd);
  pack_w_kernel<<<nWdd / 256, 256, 0, stream>>>(Wk, wk16, DD, DD, nWdd);
  pack_w_kernel<<<nWdd / 256, 256, 0, stream>>>(Wv, wv16, DD, DD, nWdd);
  pack_w_kernel<<<nWdd / 256, 256, 0, stream>>>(Wo, wo16, DD, DD, nWdd);
  pack_w_kernel<<<nWff / 256, 256, 0, stream>>>(W1, w116, DD, FFD, nWff);
  pack_w_kernel<<<nWff / 256, 256, 0, stream>>>(W2, w216, FFD, DD, nWff);

  // ---- x = rows (+ train target embeddings on first `split` columns) ----
  add_embed_kernel<<<(MM * DD) / 256, 256, 0, stream>>>(rows, tte, split_p, x32, x16);

  const int gemmBlkD  = ((MM / 16) * (DD  / 64)) / 8;  // 256 blocks
  const int gemmBlkFF = ((MM / 16) * (FFD / 64)) / 8;  // 1024 blocks
  const int attnBlk   = (BB * HH * (TT / 16)) / 8;     // 512 blocks
  const int lnBlk     = MM / 8;                        // 1024 blocks

  for (int l = 0; l < LL; ++l) {
    // QKV projections (V written transposed for attention)
    gemm_wmma_kernel<<<gemmBlkD, 256, 0, stream>>>(
        x16, wq16 + (size_t)l * DD * DD, bq + l * DD,
        nullptr, q16, nullptr, nullptr, MM, DD, DD, 0);
    gemm_wmma_kernel<<<gemmBlkD, 256, 0, stream>>>(
        x16, wk16 + (size_t)l * DD * DD, bk + l * DD,
        nullptr, k16, nullptr, nullptr, MM, DD, DD, 0);
    gemm_wmma_kernel<<<gemmBlkD, 256, 0, stream>>>(
        x16, wv16 + (size_t)l * DD * DD, bv + l * DD,
        nullptr, nullptr, nullptr, vt16, MM, DD, DD, 0);

    // masked multi-head attention (flash-style)
    attn_kernel<<<attnBlk, 256, 0, stream>>>(q16, k16, vt16, split_p, o16);

    // output projection + residual, then LN1
    gemm_wmma_kernel<<<gemmBlkD, 256, 0, stream>>>(
        o16, wo16 + (size_t)l * DD * DD, bo + l * DD,
        x32, nullptr, res32, nullptr, MM, DD, DD, 0);
    ln_kernel<<<lnBlk, 256, 0, stream>>>(res32, ln1s + l * DD, ln1b + l * DD, x32, x16);

    // FFN: h = gelu(x @ W1 + b1); resid = x + h @ W2 + b2; LN2
    gemm_wmma_kernel<<<gemmBlkFF, 256, 0, stream>>>(
        x16, w116 + (size_t)l * DD * FFD, b1 + l * FFD,
        nullptr, h16, nullptr, nullptr, MM, FFD, DD, 1);
    gemm_wmma_kernel<<<gemmBlkD, 256, 0, stream>>>(
        h16, w216 + (size_t)l * FFD * DD, b2 + l * DD,
        x32, nullptr, res32, nullptr, MM, DD, FFD, 0);
    ln_kernel<<<lnBlk, 256, 0, stream>>>(res32, ln2s + l * DD, ln2b + l * DD, x32, x16);
  }

  copy_out_kernel<<<(MM * DD) / 256, 256, 0, stream>>>(x32, (float*)d_out, MM * DD);
}
